// MultiheadAttention_24498493456448
// MI455X (gfx1250) — compile-verified, matrix operand order checked by reading, not run on hardware
//
#include <hip/hip_runtime.h>
#include <hip/hip_fp16.h>

typedef _Float16 v16h __attribute__((ext_vector_type(16)));
typedef _Float16 v8h  __attribute__((ext_vector_type(8)));
typedef float    v8f  __attribute__((ext_vector_type(8)));
typedef float    v4f  __attribute__((ext_vector_type(4)));

#define BDIM 2
#define SEQ  2048
#define EMB  2048
#define NH   16
#define HD   128

static __device__ __forceinline__ v16h cat16(v8h lo, v8h hi) {
  return __builtin_shufflevector(lo, hi, 0,1,2,3,4,5,6,7,8,9,10,11,12,13,14,15);
}

static __device__ __forceinline__ v8f wmma_f16(v16h a, v16h b, v8f c) {
  // D = A(16x32 f16) x B(32x16 f16) + C(16x16 f32)
  return __builtin_amdgcn_wmma_f32_16x16x32_f16(false, a, false, b, (short)0, c,
                                                false, false);
}

// 16-lane butterfly reductions via ds_swizzle (group-of-32 mode, imm pattern:
// xor_mask[14:10] | or_mask[9:5]=0 | and_mask[4:0]=0x1f). XOR masks 1,2,4,8
// stay inside each 16-lane half-group (WMMA C-layout row groups).
static __device__ __forceinline__ float red_max16(float x) {
  x = fmaxf(x, __int_as_float(__builtin_amdgcn_ds_swizzle(__float_as_int(x), 0x041F)));
  x = fmaxf(x, __int_as_float(__builtin_amdgcn_ds_swizzle(__float_as_int(x), 0x081F)));
  x = fmaxf(x, __int_as_float(__builtin_amdgcn_ds_swizzle(__float_as_int(x), 0x101F)));
  x = fmaxf(x, __int_as_float(__builtin_amdgcn_ds_swizzle(__float_as_int(x), 0x201F)));
  return x;
}
static __device__ __forceinline__ float red_sum16(float x) {
  x += __int_as_float(__builtin_amdgcn_ds_swizzle(__float_as_int(x), 0x041F));
  x += __int_as_float(__builtin_amdgcn_ds_swizzle(__float_as_int(x), 0x081F));
  x += __int_as_float(__builtin_amdgcn_ds_swizzle(__float_as_int(x), 0x101F));
  x += __int_as_float(__builtin_amdgcn_ds_swizzle(__float_as_int(x), 0x201F));
  return x;
}

// ---------------------------------------------------------------------------
// Tiled GEMM: C[M,N] = A[M,K] * W[K,N] (* scale). f16 WMMA, f32 accum.
// Block tile 128x128, BK=32, double-buffered LDS, 1 barrier per K-step,
// global prefetch of next tiles overlapped with WMMA compute.
// 256 threads = 8 waves (4x2); each wave: 32x64 tile = 2x4 WMMA accumulators.
// ---------------------------------------------------------------------------
template <typename TIN, typename TOUT>
__global__ __launch_bounds__(256)
void gemm_wmma(const TIN* __restrict__ A, const float* __restrict__ W,
               TOUT* __restrict__ C, int M, int N, int K, float scale)
{
  __shared__ alignas(16) _Float16 sA[2][128][40];   // [buf][m][k]
  __shared__ alignas(16) _Float16 sB[2][128][40];   // [buf][n][k] (W transposed)

  const int tid   = threadIdx.x;
  const int lane  = tid & 31;
  const int wave  = tid >> 5;
  const int wm    = wave >> 1;          // 0..3
  const int wn    = wave & 1;           // 0..1
  const int mfrag = lane & 15;
  const int kb    = (lane >> 4) << 3;   // 0 or 8

  const int gm0 = blockIdx.y * 128;
  const int gn0 = blockIdx.x * 128;

  const int a_c  = (tid & 3) * 8;       // A: col group of 8
  const int a_r  = tid >> 2;            // A: row 0..63 (+64)
  const int w_cg = (tid & 31) * 4;      // W: col group of 4
  const int w_r  = tid >> 5;            // W: k-row 0..7 (+8,16,24)

  auto loadA = [&](int k0, v8h out[2]) {
    #pragma unroll
    for (int rr = 0; rr < 2; ++rr) {
      const TIN* src = A + (size_t)(gm0 + a_r + rr * 64) * K + k0 + a_c;
      if constexpr (sizeof(TIN) == 4) {
        v4f f0 = *(const v4f*)src;
        v4f f1 = *(const v4f*)(src + 4);
        v8h h;
        #pragma unroll
        for (int j = 0; j < 4; ++j) { h[j] = (_Float16)f0[j]; h[4 + j] = (_Float16)f1[j]; }
        out[rr] = h;
      } else {
        out[rr] = *(const v8h*)src;
      }
    }
  };
  auto storeA = [&](int buf, const v8h in[2]) {
    #pragma unroll
    for (int rr = 0; rr < 2; ++rr)
      *(v8h*)&sA[buf][a_r + rr * 64][a_c] = in[rr];
  };
  auto loadW = [&](int k0, v4f out[4]) {
    #pragma unroll
    for (int r4 = 0; r4 < 4; ++r4)
      out[r4] = *(const v4f*)(W + (size_t)(k0 + w_r + r4 * 8) * N + gn0 + w_cg);
  };
  auto storeW = [&](int buf, const v4f in[4]) {
    #pragma unroll
    for (int r4 = 0; r4 < 4; ++r4)
      #pragma unroll
      for (int j = 0; j < 4; ++j)
        sB[buf][w_cg + j][w_r + r4 * 8] = (_Float16)in[r4][j];
  };

  v8f acc[2][4];
  #pragma unroll
  for (int fm = 0; fm < 2; ++fm)
    #pragma unroll
    for (int fn = 0; fn < 4; ++fn)
      acc[fm][fn] = 0.0f;

  v8h ra[2];
  v4f rw[4];
  loadA(0, ra); loadW(0, rw);
  storeA(0, ra); storeW(0, rw);

  const int NK = K / 32;
  for (int it = 0; it < NK; ++it) {
    __syncthreads();                      // buf `cur` ready (all waves)
    const int cur = it & 1, nxt = cur ^ 1;
    const bool more = (it + 1) < NK;
    if (more) { loadA((it + 1) * 32, ra); loadW((it + 1) * 32, rw); }

    v16h af[2];
    #pragma unroll
    for (int fm = 0; fm < 2; ++fm) {
      const int row = wm * 32 + fm * 16 + mfrag;
      af[fm] = cat16(*(const v8h*)&sA[cur][row][kb],
                     *(const v8h*)&sA[cur][row][kb + 16]);
    }
    #pragma unroll
    for (int fn = 0; fn < 4; ++fn) {
      const int col = wn * 64 + fn * 16 + mfrag;
      v16h bf = cat16(*(const v8h*)&sB[cur][col][kb],
                      *(const v8h*)&sB[cur][col][kb + 16]);
      #pragma unroll
      for (int fm = 0; fm < 2; ++fm)
        acc[fm][fn] = wmma_f16(af[fm], bf, acc[fm][fn]);
    }

    if (more) { storeA(nxt, ra); storeW(nxt, rw); }
  }

  const int nlane = lane & 15;
  const int mbase = (lane >> 4) * 8;
  #pragma unroll
  for (int fm = 0; fm < 2; ++fm)
    #pragma unroll
    for (int fn = 0; fn < 4; ++fn) {
      const int gcol = gn0 + wn * 64 + fn * 16 + nlane;
      #pragma unroll
      for (int r = 0; r < 8; ++r) {
        const int grow = gm0 + wm * 32 + fm * 16 + mbase + r;
        C[(size_t)grow * N + gcol] = (TOUT)(acc[fm][fn][r] * scale);
      }
    }
}

// ---------------------------------------------------------------------------
// Flash attention: grid (S/128, H, B), 256 threads. K-tiles DMA'd into LDS
// with global_load_async_to_lds_b128 (ASYNCcnt); V-tiles transposed through
// VGPRs; double-buffered, 1 workgroup barrier per 32-key step.
// ---------------------------------------------------------------------------
__global__ __launch_bounds__(256)
void attn_wmma(const _Float16* __restrict__ Q, const _Float16* __restrict__ Kh,
               const _Float16* __restrict__ V, _Float16* __restrict__ O)
{
  __shared__ alignas(16) _Float16 sK[2][32][136];    // [buf][key][d]
  __shared__ alignas(16) _Float16 sVt[2][128][40];   // [buf][d][key]
  __shared__ alignas(16) _Float16 sP[8][16][32];     // per-wave probs [m][key]

  const int tid   = threadIdx.x;
  const int lane  = tid & 31;
  const int wave  = tid >> 5;
  const int qt    = blockIdx.x;
  const int h     = blockIdx.y;
  const int b     = blockIdx.z;
  const int mfrag = lane & 15;
  const int kb    = (lane >> 4) << 3;
  const int hoff  = h * HD;

  // --- async DMA of K tile: 512 x 16B chunks, 2 per thread ---
  auto issueK = [&](int kt, int buf) {
    #pragma unroll
    for (int i = 0; i < 2; ++i) {
      const int c   = tid * 2 + i;        // 0..511
      const int key = c >> 4;
      const int chk = c & 15;
      const _Float16* gp =
          Kh + ((size_t)b * SEQ + kt + key) * EMB + hoff + chk * 8;
      unsigned lds = (unsigned)(size_t)&sK[buf][key][chk * 8];
      unsigned long long ga = (unsigned long long)(size_t)gp;
      asm volatile("global_load_async_to_lds_b128 %0, %1, off"
                   :: "v"(lds), "v"(ga) : "memory");
    }
  };
  // --- V tile: load 16 d-values for one key, store transposed ---
  const int v_key = tid & 31;
  const int v_dg  = (tid >> 5) * 16;
  auto loadV = [&](int kt, v8h out[2]) {
    const _Float16* vp =
        V + ((size_t)b * SEQ + kt + v_key) * EMB + hoff + v_dg;
    out[0] = *(const v8h*)vp;
    out[1] = *(const v8h*)(vp + 8);
  };
  auto storeV = [&](int buf, const v8h in[2]) {
    #pragma unroll
    for (int j = 0; j < 8; ++j) {
      sVt[buf][v_dg + j][v_key]     = in[0][j];
      sVt[buf][v_dg + 8 + j][v_key] = in[1][j];
    }
  };

  // Q A-fragments (Q pre-scaled by 1/sqrt(D))
  const size_t qrow = (size_t)b * SEQ + (size_t)qt * 128 + wave * 16 + mfrag;
  v16h aq[4];
  #pragma unroll
  for (int j = 0; j < 4; ++j) {
    const _Float16* p = Q + qrow * EMB + hoff + j * 32 + kb;
    aq[j] = cat16(*(const v8h*)p, *(const v8h*)(p + 16));
  }

  v8f acc[8];
  float mrow[8], lrow[8];
  #pragma unroll
  for (int jd = 0; jd < 8; ++jd) acc[jd] = 0.0f;
  #pragma unroll
  for (int r = 0; r < 8; ++r) { mrow[r] = -3.0e38f; lrow[r] = 0.0f; }

  const float LOG2E = 1.44269504088896340736f;

  v8h rv[2];
  issueK(0, 0);
  loadV(0, rv);
  storeV(0, rv);

  const int NKT = SEQ / 32;
  for (int it = 0; it < NKT; ++it) {
    asm volatile("s_wait_asynccnt 0x0" ::: "memory");  // my K DMA landed
    __syncthreads();                                    // everyone's tiles ready
    const int cur = it & 1, nxt = cur ^ 1;
    const bool more = (it + 1) < NKT;
    if (more) { issueK((it + 1) * 32, nxt); loadV((it + 1) * 32, rv); }

    // ---- S = Q * K^T (two 16x16 key tiles) ----
    v8f s0 = 0.0f, s1 = 0.0f;
    #pragma unroll
    for (int j = 0; j < 4; ++j) {
      const _Float16* pk0 = &sK[cur][mfrag][j * 32 + kb];
      const _Float16* pk1 = &sK[cur][16 + mfrag][j * 32 + kb];
      v16h b0 = cat16(*(const v8h*)pk0, *(const v8h*)(pk0 + 16));
      v16h b1 = cat16(*(const v8h*)pk1, *(const v8h*)(pk1 + 16));
      s0 = wmma_f16(aq[j], b0, s0);
      s1 = wmma_f16(aq[j], b1, s1);
    }

    // ---- online softmax (rows in 16-lane half-groups) ----
    float alpha[8];
    #pragma unroll
    for (int r = 0; r < 8; ++r) {
      const float mx   = red_max16(fmaxf(s0[r], s1[r]));
      const float mnew = fmaxf(mrow[r], mx);
      alpha[r] = exp2f((mrow[r] - mnew) * LOG2E);
      const float p0 = exp2f((s0[r] - mnew) * LOG2E);
      const float p1 = exp2f((s1[r] - mnew) * LOG2E);
      lrow[r] = lrow[r] * alpha[r] + red_sum16(p0 + p1);
      mrow[r] = mnew;
      const int m = ((lane >> 4) << 3) + r;
      sP[wave][m][mfrag]      = (_Float16)p0;
      sP[wave][m][16 + mfrag] = (_Float16)p1;
    }
    #pragma unroll
    for (int jd = 0; jd < 8; ++jd)
      #pragma unroll
      for (int r = 0; r < 8; ++r)
        acc[jd][r] *= alpha[r];

    // wave-local: P writes are only read by this wave (LDS is in-order/wave)
    asm volatile("s_wait_dscnt 0x0" ::: "memory");
    __builtin_amdgcn_wave_barrier();

    // ---- ctx += P * V ----
    const _Float16* pp = &sP[wave][mfrag][kb];
    v16h ap = cat16(*(const v8h*)pp, *(const v8h*)(pp + 16));
    #pragma unroll
    for (int jd = 0; jd < 8; ++jd) {
      const _Float16* pv = &sVt[cur][jd * 16 + mfrag][kb];
      v16h bv = cat16(*(const v8h*)pv, *(const v8h*)(pv + 16));
      acc[jd] = wmma_f16(ap, bv, acc[jd]);
    }

    if (more) storeV(nxt, rv);   // drain V prefetch into the other buffer
  }

  // ---- epilogue: divide by row sums, store ctx (f16) ----
  const int mbase = (lane >> 4) * 8;
  float inv[8];
  #pragma unroll
  for (int r = 0; r < 8; ++r) inv[r] = 1.0f / lrow[r];
  #pragma unroll
  for (int jd = 0; jd < 8; ++jd) {
    const int d = jd * 16 + mfrag;
    #pragma unroll
    for (int r = 0; r < 8; ++r) {
      const size_t orow =
          (size_t)b * SEQ + (size_t)qt * 128 + wave * 16 + mbase + r;
      O[orow * EMB + hoff + d] = (_Float16)(acc[jd][r] * inv[r]);
    }
  }
}

// ---------------------------------------------------------------------------
extern "C" void kernel_launch(void* const* d_in, const int* in_sizes, int n_in,
                              void* d_out, int out_size, void* d_ws, size_t ws_size,
                              hipStream_t stream)
{
  const float* q  = (const float*)d_in[0];
  const float* k  = (const float*)d_in[1];
  const float* v  = (const float*)d_in[2];
  const float* wq = (const float*)d_in[3];
  const float* wk = (const float*)d_in[4];
  const float* wv = (const float*)d_in[5];
  const float* wo = (const float*)d_in[6];

  const size_t nelem = (size_t)BDIM * SEQ * EMB;   // 8M halfs per buffer
  _Float16* Qh = (_Float16*)d_ws;
  _Float16* Kh = Qh + nelem;
  _Float16* Vh = Kh + nelem;
  _Float16* Ch = Vh + nelem;

  const int M = BDIM * SEQ, N = EMB, K = EMB;
  dim3 blk(256);
  dim3 g1(N / 128, M / 128);
  const float qscale = 0.08838834764831845f;       // 1/sqrt(128)

  gemm_wmma<float, _Float16><<<g1, blk, 0, stream>>>(q, wq, Qh, M, N, K, qscale);
  gemm_wmma<float, _Float16><<<g1, blk, 0, stream>>>(k, wk, Kh, M, N, K, 1.0f);
  gemm_wmma<float, _Float16><<<g1, blk, 0, stream>>>(v, wv, Vh, M, N, K, 1.0f);

  dim3 ga(SEQ / 128, NH, BDIM);
  attn_wmma<<<ga, blk, 0, stream>>>(Qh, Kh, Vh, Ch);

  gemm_wmma<_Float16, float><<<g1, blk, 0, stream>>>(Ch, wo, (float*)d_out, M, N, K, 1.0f);
}
